// QKVAttention_encoder_83502754169208
// MI455X (gfx1250) — matrix operand order repaired and checked by reading, not compiled, not run on hardware
//
#include <hip/hip_runtime.h>
#include <hip/hip_bf16.h>

typedef __attribute__((ext_vector_type(16))) __bf16 v16bf;
typedef __attribute__((ext_vector_type(8)))  __bf16 v8bf;
typedef __attribute__((ext_vector_type(8)))  float  v8f;

namespace {

constexpr int kHeads = 8;
constexpr int kComp  = 256;
constexpr int kTok   = 1024;
constexpr int kCh    = 128;

// ---- LDS layout (byte offsets), total 142336 B (< 320 KB/WGP) ----
constexpr int QCT_OFF   = 0;        // [256][136] bf16 : Qc^T, rows o (compress), cols c
constexpr int KT_OFF    = 69632;    // [ 64][136] bf16 : K^T for s-block, rows s, cols c
constexpr int V_OFF     = 87040;    // [128][ 72] bf16 : V for s-block, rows c, cols s
constexpr int P_OFF     = 105472;   // [256][ 72] bf16 : probs, rows o, cols s
constexpr int WQS_OFF   = 69632;    // phase1 staging: Wq chunk [256][40] bf16 (aliases KT/V/P)
constexpr int QS_OFF    = 90112;    // phase1 staging: Q  chunk [128][40] bf16
constexpr int OUT_OFF   = 0;        // final: O transpose buffer [128][260] f32 (aliases QCT)
constexpr int LDS_BYTES = 142336;

__device__ __forceinline__ v8f wmma_bf16(v16bf a, v16bf b, v8f c) {
  // D(16x16 f32) = A(16x32 bf16) x B(32x16 bf16) + C
  return __builtin_amdgcn_wmma_f32_16x16x32_bf16(false, a, false, b, (short)0, c,
                                                 false, false);
}

// A-fragment (M x K): LDS array is row-major [M-rows][K-cols], stride in bf16 elems.
// ISA layout: lane L -> M = row + (L&15); elem e -> K = col + 8*(L>=16) + e + (e>=8 ? 8 : 0)
__device__ __forceinline__ v16bf frag_a(const __bf16* base, int row, int stride,
                                        int col, int l15, int hf) {
  const __bf16* p = base + (row + l15) * stride + col + hf * 8;
  v8bf lo = *(const v8bf*)(p);
  v8bf hi = *(const v8bf*)(p + 16);
  return __builtin_shufflevector(lo, hi, 0, 1, 2, 3, 4, 5, 6, 7,
                                 8, 9, 10, 11, 12, 13, 14, 15);
}

// B-fragment (K x N): LDS array is row-major [N-rows][K-cols].
// ISA layout: lane L -> N = row + (L&15); elem e -> K = col + 8*(L>=16) + e + (e>=8 ? 8 : 0)
__device__ __forceinline__ v16bf frag_b(const __bf16* base, int row, int stride,
                                        int col, int l15, int hf) {
  const __bf16* p = base + (row + l15) * stride + col + hf * 8;
  v8bf lo = *(const v8bf*)(p);
  v8bf hi = *(const v8bf*)(p + 16);
  return __builtin_shufflevector(lo, hi, 0, 1, 2, 3, 4, 5, 6, 7,
                                 8, 9, 10, 11, 12, 13, 14, 15);
}

// Reductions across the 16-lane groups of a wave32 (D-tile halves stay separate).
__device__ __forceinline__ float redmax16(float x) {
#pragma unroll
  for (int m = 1; m < 16; m <<= 1) x = fmaxf(x, __shfl_xor(x, m, 32));
  return x;
}
__device__ __forceinline__ float redsum16(float x) {
#pragma unroll
  for (int m = 1; m < 16; m <<= 1) x += __shfl_xor(x, m, 32);
  return x;
}

}  // namespace

__global__ __launch_bounds__(512)
void qkv_attn_flash(const float* __restrict__ qkv, const float* __restrict__ w_q,
                    const float* __restrict__ b_q, float* __restrict__ out) {
  extern __shared__ char smem[];
  __bf16* qct = (__bf16*)(smem + QCT_OFF);
  __bf16* kts = (__bf16*)(smem + KT_OFF);
  __bf16* vsh = (__bf16*)(smem + V_OFF);
  __bf16* psh = (__bf16*)(smem + P_OFF);
  __bf16* wqs = (__bf16*)(smem + WQS_OFF);
  __bf16* qsh = (__bf16*)(smem + QS_OFF);
  float*  osh = (float*)(smem + OUT_OFF);

  const int tid  = threadIdx.x;
  const int wv   = tid >> 5;        // wave id 0..15 -> owns compress rows [16*wv, 16*wv+16)
  const int lane = tid & 31;
  const int l15  = lane & 15;
  const int hf   = lane >> 4;

  const int bh    = blockIdx.x;     // 0..255 (batch*heads)
  const int batch = bh >> 3;
  const int head  = bh & 7;
  const long qb   = (long)batch * (3 * kHeads * kCh) * kTok + (long)head * (3 * kCh) * kTok;
  const float s2  = 0.08838834764831845f;  // 1/sqrt(128) == scale*scale

  v8f zero = {};

  // =====================================================================
  // Phase 1: Qc^T[o][c] = (sum_t Wq[o][t]*Q[c][t] + b[o]) * s2   (bf16 in LDS)
  //   M = o (wave-owned 16-row tile), N = c (8 tiles), K = t (32 per chunk)
  //   Staging is double-buffered in registers: loads for chunk i+1 fly
  //   under the WMMAs of chunk i.
  // =====================================================================
  // per-thread staging coordinates
  const int p1_row8 = 0;  // dummy to keep structure clear
  (void)p1_row8;

  float4 wq_buf[4];
  float4 q_buf[2];

  auto load_phase1 = [&](int tBase) {
#pragma unroll
    for (int j = 0; j < 4; ++j) {
      int slot = tid + 512 * j;
      int row  = slot >> 3;
      int q4   = (slot & 7) * 4;
      wq_buf[j] = *(const float4*)(w_q + (long)row * kTok + tBase + q4);
    }
#pragma unroll
    for (int j = 0; j < 2; ++j) {
      int slot = tid + 512 * j;
      int row  = slot >> 3;
      int q4   = (slot & 7) * 4;
      q_buf[j] = *(const float4*)(qkv + qb + (long)row * kTok + tBase + q4);
    }
  };

  auto store_phase1 = [&]() {
#pragma unroll
    for (int j = 0; j < 4; ++j) {
      int slot = tid + 512 * j;
      int row  = slot >> 3;
      int q4   = (slot & 7) * 4;
      __bf16* d = wqs + row * 40 + q4;
      d[0] = (__bf16)wq_buf[j].x; d[1] = (__bf16)wq_buf[j].y;
      d[2] = (__bf16)wq_buf[j].z; d[3] = (__bf16)wq_buf[j].w;
    }
#pragma unroll
    for (int j = 0; j < 2; ++j) {
      int slot = tid + 512 * j;
      int row  = slot >> 3;
      int q4   = (slot & 7) * 4;
      __bf16* d = qsh + row * 40 + q4;
      d[0] = (__bf16)q_buf[j].x; d[1] = (__bf16)q_buf[j].y;
      d[2] = (__bf16)q_buf[j].z; d[3] = (__bf16)q_buf[j].w;
    }
  };

  v8f accQ[8];
#pragma unroll
  for (int n = 0; n < 8; ++n) accQ[n] = zero;

  load_phase1(0);
  for (int kt = 0; kt < kTok / 32; ++kt) {
    __syncthreads();       // previous chunk's fragment reads complete
    store_phase1();
    if (kt + 1 < kTok / 32) load_phase1((kt + 1) * 32);  // prefetch next chunk
    __syncthreads();

    v16bf a = frag_a(wqs, 16 * wv, 40, 0, l15, hf);
#pragma unroll
    for (int n = 0; n < 8; ++n) {
      v16bf b = frag_b(qsh, 16 * n, 40, 0, l15, hf);
      accQ[n] = wmma_bf16(a, b, accQ[n]);
    }
  }

  // bias + scale, write Qc^T (each wave writes only its own rows -> no barrier)
  float bq[8];
#pragma unroll
  for (int v = 0; v < 8; ++v) bq[v] = b_q[16 * wv + v + 8 * hf];
#pragma unroll
  for (int n = 0; n < 8; ++n) {
#pragma unroll
    for (int v = 0; v < 8; ++v) {
      int c = 16 * n + l15;
      int o = 16 * wv + v + 8 * hf;
      qct[o * 136 + c] = (__bf16)((accQ[n][v] + bq[v]) * s2);
    }
  }

  // =====================================================================
  // Phase 2+3: flash attention over s-blocks of 64 tokens,
  // register-double-buffered K/V staging.
  // =====================================================================
  float4 k_buf[4], v_buf[4];

  auto load_kv = [&](int sBase) {
#pragma unroll
    for (int j = 0; j < 4; ++j) {
      int slot = tid + 512 * j;
      int c    = slot >> 4;
      int s0   = (slot & 15) * 4;
      k_buf[j] = *(const float4*)(qkv + qb + (long)(kCh + c) * kTok + sBase + s0);
      v_buf[j] = *(const float4*)(qkv + qb + (long)(2 * kCh + c) * kTok + sBase + s0);
    }
    // warm L2 for the block after next (emits global_prefetch_b8)
    if (sBase + 64 < kTok) {
      int c0 = tid >> 4;
      int s0 = (tid & 15) * 4;
      __builtin_prefetch(qkv + qb + (long)(kCh + c0) * kTok + sBase + 64 + s0, 0, 1);
      __builtin_prefetch(qkv + qb + (long)(2 * kCh + c0) * kTok + sBase + 64 + s0, 0, 1);
    }
  };

  auto store_kv = [&]() {
#pragma unroll
    for (int j = 0; j < 4; ++j) {
      int slot = tid + 512 * j;
      int c    = slot >> 4;
      int s0   = (slot & 15) * 4;
      kts[(s0 + 0) * 136 + c] = (__bf16)k_buf[j].x;
      kts[(s0 + 1) * 136 + c] = (__bf16)k_buf[j].y;
      kts[(s0 + 2) * 136 + c] = (__bf16)k_buf[j].z;
      kts[(s0 + 3) * 136 + c] = (__bf16)k_buf[j].w;
      __bf16* d = vsh + c * 72 + s0;
      d[0] = (__bf16)v_buf[j].x; d[1] = (__bf16)v_buf[j].y;
      d[2] = (__bf16)v_buf[j].z; d[3] = (__bf16)v_buf[j].w;
    }
  };

  v8f accO[8];
  float m[8], l[8];
#pragma unroll
  for (int n = 0; n < 8; ++n) accO[n] = zero;
#pragma unroll
  for (int v = 0; v < 8; ++v) { m[v] = -1.0e30f; l[v] = 0.0f; }

  load_kv(0);
  for (int sb = 0; sb < kTok / 64; ++sb) {
    __syncthreads();  // previous block's K/V/P reads complete (also fences phase-1 alias)
    store_kv();
    if (sb + 1 < kTok / 64) load_kv((sb + 1) * 64);  // next block flies under compute
    __syncthreads();

    // --- S = Qc^T x K : M=o (wave tile), N=s (4 tiles), K=c (4 chunks) ---
    v8f S[4];
#pragma unroll
    for (int n = 0; n < 4; ++n) S[n] = zero;
    for (int kc = 0; kc < 4; ++kc) {
      v16bf a = frag_a(qct, 16 * wv, 136, kc * 32, l15, hf);
#pragma unroll
      for (int n = 0; n < 4; ++n) {
        v16bf b = frag_b(kts, 16 * n, 136, kc * 32, l15, hf);
        S[n] = wmma_bf16(a, b, S[n]);
      }
    }

    // --- online softmax per row o = 16*wv + v + 8*hf ---
#pragma unroll
    for (int v = 0; v < 8; ++v) {
      float mv = S[0][v];
#pragma unroll
      for (int n = 1; n < 4; ++n) mv = fmaxf(mv, S[n][v]);
      mv = redmax16(mv);
      float mn = fmaxf(m[v], mv);
      float fac = __expf(m[v] - mn);
      m[v] = mn;
      float rs = 0.0f;
#pragma unroll
      for (int n = 0; n < 4; ++n) {
        float p = __expf(S[n][v] - mn);
        S[n][v] = p;
        rs += p;
      }
      rs = redsum16(rs);
      l[v] = l[v] * fac + rs;
#pragma unroll
      for (int n2 = 0; n2 < 8; ++n2) accO[n2][v] *= fac;
      // write P row (only this wave reads it back; LDS same-wave ordering suffices)
      int o = 16 * wv + v + 8 * hf;
#pragma unroll
      for (int n = 0; n < 4; ++n) psh[o * 72 + 16 * n + l15] = (__bf16)S[n][v];
    }

    // --- O += P x V^T : M=o (wave tile), N=c (8 tiles), K=s (2 chunks) ---
    for (int ks = 0; ks < 2; ++ks) {
      v16bf a = frag_a(psh, 16 * wv, 72, ks * 32, l15, hf);
#pragma unroll
      for (int n2 = 0; n2 < 8; ++n2) {
        v16bf b = frag_b(vsh, 16 * n2, 72, ks * 32, l15, hf);
        accO[n2] = wmma_bf16(a, b, accO[n2]);
      }
    }
  }

  // =====================================================================
  // Epilogue: normalize, transpose through LDS, coalesced f32 store
  //   out[batch][head*128 + c][o], inner dim o (256) contiguous
  // =====================================================================
  float inv[8];
#pragma unroll
  for (int v = 0; v < 8; ++v) inv[v] = 1.0f / l[v];

  __syncthreads();  // all waves done reading qct before we alias it
#pragma unroll
  for (int n = 0; n < 8; ++n) {
#pragma unroll
    for (int v = 0; v < 8; ++v) {
      int c = 16 * n + l15;
      int o = 16 * wv + v + 8 * hf;
      osh[c * 260 + o] = accO[n][v] * inv[v];
    }
  }
  __syncthreads();

  const long ob = (long)batch * (kHeads * kCh) * kComp + (long)head * kCh * kComp;
#pragma unroll
  for (int j = 0; j < 16; ++j) {
    int slot = tid + 512 * j;
    int flat = slot * 4;
    int c = flat >> 8;
    int o = flat & 255;
    float4 x = *(const float4*)(osh + c * 260 + o);
    *(float4*)(out + ob + (long)c * kComp + o) = x;
  }
}

extern "C" void kernel_launch(void* const* d_in, const int* in_sizes, int n_in,
                              void* d_out, int out_size, void* d_ws, size_t ws_size,
                              hipStream_t stream) {
  (void)in_sizes; (void)n_in; (void)d_ws; (void)ws_size; (void)out_size;
  const float* qkv = (const float*)d_in[0];
  const float* w_q = (const float*)d_in[1];
  const float* b_q = (const float*)d_in[2];
  float* out = (float*)d_out;

  // Allow > 64 KB dynamic LDS (gfx1250 WGP has 320 KB). Deterministic, idempotent.
  (void)hipFuncSetAttribute((const void*)qkv_attn_flash,
                            hipFuncAttributeMaxDynamicSharedMemorySize, LDS_BYTES);

  qkv_attn_flash<<<dim3(32 * kHeads), dim3(512), LDS_BYTES, stream>>>(qkv, w_q, b_q, out);
}
